// ProteinMPNNFMIF_80264348827950
// MI455X (gfx1250) — matrix operand order repaired and checked by reading, not compiled, not run on hardware
//
#include <hip/hip_runtime.h>
#include <math.h>

typedef __attribute__((ext_vector_type(16))) __bf16 v16bf;
typedef __attribute__((ext_vector_type(8)))  float  v8f;

#define NB 2
#define NN 1024
#define NKN 48
#define NH 128
#define H3 384
#define FF 512

__device__ __forceinline__ float gelu_exact(float x) {
    return 0.5f * x * (1.0f + erff(x * 0.70710678118654752440f));
}

__device__ __forceinline__ v8f wmma_bf16(v16bf a, v16bf b, v8f c) {
    return __builtin_amdgcn_wmma_f32_16x16x32_bf16(false, a, false, b, (short)0, c, false, false);
}

// A fragment: 16x32 bf16 tile from an LDS buffer (row-major, element stride `stride`).
// Contiguous k order used here (not the ISA table order): lanes 0-15 / 16-31 both hold M = lane&15; VGPR v holds packed K pair
// K = (lane>>4)*16 + v*2.
__device__ __forceinline__ v16bf load_a_frag(const __bf16* base, int stride, int row0, int k0) {
    const int lane = threadIdx.x & 31;
    const int g = lane >> 4, m = lane & 15;
    const __bf16* p = base + (row0 + m) * stride + k0;
    v16bf a;
#pragma unroll
    for (int v = 0; v < 8; ++v) {
        const int k = (g << 4) + (v << 1);
        union { unsigned u; __bf16 h[2]; } cv;
        cv.u = *(const unsigned*)(p + k);
        a[2 * v] = cv.h[0];
        a[2 * v + 1] = cv.h[1];
    }
    return a;
}

// B fragment: 32x16 (KxN) from transposed bf16 weights Wt[N][K] (leading dim ldk).
// lanes 0-15: K=0..15 (packed pairs per VGPR), lanes 16-31: K=16..31; N = lane&15.
__device__ __forceinline__ v16bf load_b_frag(const __bf16* Wt, int ldk, int col0, int k0) {
    const int lane = threadIdx.x & 31;
    const int g = lane >> 4, n = lane & 15;
    const __bf16* p = Wt + (size_t)(col0 + n) * ldk + k0 + (g << 4);
    v16bf b;
#pragma unroll
    for (int v = 0; v < 8; ++v) {
        union { unsigned u; __bf16 h[2]; } cv;
        cv.u = *(const unsigned*)(p + 2 * v);
        b[2 * v] = cv.h[0];
        b[2 * v + 1] = cv.h[1];
    }
    return b;
}

// Store 16x16 f32 accumulator tile (+bias, GELU) as bf16 into LDS.
// D layout: VGPR r -> row = r + 8*(lane>>4); col = lane&15.
__device__ __forceinline__ void store_gelu_tile(__bf16* dst, int stride, int row0, int col0,
                                                v8f acc, const float* bias) {
    const int lane = threadIdx.x & 31;
    const int g = lane >> 4, c = lane & 15;
    const float bv = bias[col0 + c];
#pragma unroll
    for (int r = 0; r < 8; ++r) {
        const float v = gelu_exact(acc[r] + bv);
        dst[(row0 + r + (g << 3)) * stride + col0 + c] = (__bf16)v;
    }
}

// Convert fp32 weights [3][Kd][Nd] -> transposed bf16 [3][Nd][Kd]
__global__ void convert_transpose(const float* __restrict__ src, __bf16* __restrict__ dst,
                                  int Kd, int Nd) {
    const int total = 3 * Kd * Nd;
    for (int i = blockIdx.x * blockDim.x + threadIdx.x; i < total; i += gridDim.x * blockDim.x) {
        const int l = i / (Kd * Nd);
        const int rem = i - l * (Kd * Nd);
        const int k = rem / Nd, n = rem - k * Nd;
        dst[(size_t)l * Kd * Nd + (size_t)n * Kd + k] = (__bf16)src[i];
    }
}

// One workgroup per node (b,n). 8 waves. Computes the 3-layer message MLP for the
// 48 neighbor rows with WMMA, then either (mode 0) masked-sum aggregate + residual +
// LayerNorm into `outp` (h_V rows), or (mode 1) per-edge residual + LayerNorm in place.
__global__ void __launch_bounds__(256) msg_kernel(
    const float* hv, const float* he,
    const int* __restrict__ eidx, const float* __restrict__ mask_att,
    const __bf16* __restrict__ W1t, const float* __restrict__ b1,
    const __bf16* __restrict__ W2t, const float* __restrict__ b2,
    const __bf16* __restrict__ W3t, const float* __restrict__ b3,
    const float* __restrict__ gam, const float* __restrict__ bet,
    float* outp, int mode)
{
    __shared__ __align__(16) __bf16 sEV[NKN * 392];   // 48x384 h_EV tile (pad 8)
    __shared__ __align__(16) __bf16 sMA[NKN * 136];   // 48x128 m1
    __shared__ __align__(16) __bf16 sMB[NKN * 136];   // 48x128 m2
    __shared__ float sX[NH];
    __shared__ float sStats[2];
    float* m3 = reinterpret_cast<float*>(sEV);        // 48x132 f32, aliases sEV (dead by then)

    const int bn = blockIdx.x;
    const int b = bn >> 10;
    const int tid = threadIdx.x;
    const int wave = tid >> 5, lane = tid & 31;

    const size_t hv_base = (size_t)bn * NH;
    const size_t he_base = (size_t)bn * NKN * NH;
    const int ei_base = bn * NKN;

    // ---- gather h_EV = [h_V(center) | h_E | h_V(neighbor)] as bf16 ----
    for (int e = tid; e < NKN * H3; e += 256) {
        const int k = e / H3, c = e - k * H3;
        float v;
        if (c < NH)           v = hv[hv_base + c];
        else if (c < 2 * NH)  v = he[he_base + (size_t)k * NH + (c - NH)];
        else {
            const int j = eidx[ei_base + k];
            v = hv[((size_t)b * NN + j) * NH + (c - 2 * NH)];
        }
        sEV[k * 392 + c] = (__bf16)v;
    }
    __syncthreads();

    const int col0 = wave << 4;   // each wave owns one 16-wide output column tile

    // ---- layer 1: m1 = GELU(hEV @ W1 + b1), K = 384 ----
    {
        v8f a0 = {}, a1 = {}, a2 = {};
        for (int kc = 0; kc < H3; kc += 32) {
            const v16bf bf = load_b_frag(W1t, H3, col0, kc);
            a0 = wmma_bf16(load_a_frag(sEV, 392, 0,  kc), bf, a0);
            a1 = wmma_bf16(load_a_frag(sEV, 392, 16, kc), bf, a1);
            a2 = wmma_bf16(load_a_frag(sEV, 392, 32, kc), bf, a2);
        }
        store_gelu_tile(sMA, 136, 0,  col0, a0, b1);
        store_gelu_tile(sMA, 136, 16, col0, a1, b1);
        store_gelu_tile(sMA, 136, 32, col0, a2, b1);
    }
    __syncthreads();

    // ---- layer 2: m2 = GELU(m1 @ W2 + b2), K = 128 ----
    {
        v8f a0 = {}, a1 = {}, a2 = {};
        for (int kc = 0; kc < NH; kc += 32) {
            const v16bf bf = load_b_frag(W2t, NH, col0, kc);
            a0 = wmma_bf16(load_a_frag(sMA, 136, 0,  kc), bf, a0);
            a1 = wmma_bf16(load_a_frag(sMA, 136, 16, kc), bf, a1);
            a2 = wmma_bf16(load_a_frag(sMA, 136, 32, kc), bf, a2);
        }
        store_gelu_tile(sMB, 136, 0,  col0, a0, b2);
        store_gelu_tile(sMB, 136, 16, col0, a1, b2);
        store_gelu_tile(sMB, 136, 32, col0, a2, b2);
    }
    __syncthreads();

    // ---- layer 3: m3 = m2 @ W3 + b3 (fp32 out), K = 128 ----
    {
        v8f a0 = {}, a1 = {}, a2 = {};
        for (int kc = 0; kc < NH; kc += 32) {
            const v16bf bf = load_b_frag(W3t, NH, col0, kc);
            a0 = wmma_bf16(load_a_frag(sMB, 136, 0,  kc), bf, a0);
            a1 = wmma_bf16(load_a_frag(sMB, 136, 16, kc), bf, a1);
            a2 = wmma_bf16(load_a_frag(sMB, 136, 32, kc), bf, a2);
        }
        const int g = lane >> 4, c = lane & 15;
        const float bv = b3[col0 + c];
#pragma unroll
        for (int r = 0; r < 8; ++r) {
            const int rw = r + (g << 3);
            m3[(rw)      * 132 + col0 + c] = a0[r] + bv;
            m3[(rw + 16) * 132 + col0 + c] = a1[r] + bv;
            m3[(rw + 32) * 132 + col0 + c] = a2[r] + bv;
        }
    }
    __syncthreads();

    if (mode == 0) {
        // masked sum over K, /SCALE, residual, LayerNorm -> outp (h_V row)
        if (tid < NH) {
            float s = 0.f;
            for (int k = 0; k < NKN; ++k)
                s += mask_att[ei_base + k] * m3[k * 132 + tid];
            sX[tid] = hv[hv_base + tid] + s * (1.0f / 30.0f);
        }
        __syncthreads();
        if (tid < 32) {
            float s = 0.f, s2 = 0.f;
#pragma unroll
            for (int j = 0; j < 4; ++j) { const float x = sX[tid * 4 + j]; s += x; s2 += x * x; }
#pragma unroll
            for (int m = 16; m >= 1; m >>= 1) { s += __shfl_xor(s, m); s2 += __shfl_xor(s2, m); }
            if (tid == 0) {
                const float mu = s * (1.0f / 128.f);
                sStats[0] = mu;
                sStats[1] = s2 * (1.0f / 128.f) - mu * mu;
            }
        }
        __syncthreads();
        if (tid < NH) {
            const float mu = sStats[0];
            const float rs = rsqrtf(sStats[1] + 1e-5f);
            outp[hv_base + tid] = gam[tid] * (sX[tid] - mu) * rs + bet[tid];
        }
    } else {
        // per-edge residual + LayerNorm, in place on h_E
        for (int row = wave; row < NKN; row += 8) {
            const size_t ro = he_base + (size_t)row * NH;
            const int c0 = lane * 4;
            float x[4];
            float s = 0.f, s2 = 0.f;
#pragma unroll
            for (int j = 0; j < 4; ++j) {
                x[j] = he[ro + c0 + j] + m3[row * 132 + c0 + j];
                s += x[j]; s2 += x[j] * x[j];
            }
#pragma unroll
            for (int m = 16; m >= 1; m >>= 1) { s += __shfl_xor(s, m); s2 += __shfl_xor(s2, m); }
            const float mu = s * (1.0f / 128.f);
            const float rs = rsqrtf(s2 * (1.0f / 128.f) - mu * mu + 1e-5f);
#pragma unroll
            for (int j = 0; j < 4; ++j)
                outp[ro + c0 + j] = gam[c0 + j] * (x[j] - mu) * rs + bet[c0 + j];
        }
    }
}

// Position-wise FFN + residual + LN + mask_V. One workgroup = 16 node rows.
__global__ void __launch_bounds__(256) ffn_kernel(
    const float* __restrict__ hvin, const float* __restrict__ maskV,
    const __bf16* __restrict__ Wint, const float* __restrict__ binl,
    const __bf16* __restrict__ Woutt, const float* __restrict__ boutl,
    const float* __restrict__ gam, const float* __restrict__ bet,
    float* __restrict__ hvout)
{
    __shared__ __align__(16) __bf16 sXT[16 * 136];
    __shared__ __align__(16) __bf16 sT[16 * 520];
    __shared__ float sO[16 * 132];

    const int tid = threadIdx.x, wave = tid >> 5, lane = tid & 31;
    const size_t r0 = (size_t)blockIdx.x * 16;

    for (int e = tid; e < 16 * NH; e += 256) {
        const int i = e >> 7, c = e & 127;
        sXT[i * 136 + c] = (__bf16)hvin[(r0 + i) * NH + c];
    }
    __syncthreads();

    // T = GELU(X @ Win + bin): 32 column tiles, 4 per wave; K = 128
    {
        v8f acc[4] = {{}, {}, {}, {}};
        for (int kc = 0; kc < NH; kc += 32) {
            const v16bf a = load_a_frag(sXT, 136, 0, kc);
#pragma unroll
            for (int j = 0; j < 4; ++j) {
                const int c0 = ((wave << 2) + j) << 4;
                acc[j] = wmma_bf16(a, load_b_frag(Wint, NH, c0, kc), acc[j]);
            }
        }
#pragma unroll
        for (int j = 0; j < 4; ++j)
            store_gelu_tile(sT, 520, 0, ((wave << 2) + j) << 4, acc[j], binl);
    }
    __syncthreads();

    // O = T @ Wout + bout: 8 column tiles, 1 per wave; K = 512
    {
        const int c0 = wave << 4;
        v8f acc = {};
        for (int kc = 0; kc < FF; kc += 32)
            acc = wmma_bf16(load_a_frag(sT, 520, 0, kc), load_b_frag(Woutt, FF, c0, kc), acc);
        const int g = lane >> 4, c = lane & 15;
        const float bv = boutl[c0 + c];
#pragma unroll
        for (int r = 0; r < 8; ++r)
            sO[(r + (g << 3)) * 132 + c0 + c] = acc[r] + bv;
    }
    __syncthreads();

    // residual + LN + mask_V, two rows per wave
#pragma unroll
    for (int rr = 0; rr < 2; ++rr) {
        const int row = wave + rr * 8;
        const size_t gr = r0 + row;
        const int c0 = lane * 4;
        float x[4];
        float s = 0.f, s2 = 0.f;
#pragma unroll
        for (int j = 0; j < 4; ++j) {
            x[j] = hvin[gr * NH + c0 + j] + sO[row * 132 + c0 + j];
            s += x[j]; s2 += x[j] * x[j];
        }
#pragma unroll
        for (int m = 16; m >= 1; m >>= 1) { s += __shfl_xor(s, m); s2 += __shfl_xor(s2, m); }
        const float mu = s * (1.0f / 128.f);
        const float rs = rsqrtf(s2 * (1.0f / 128.f) - mu * mu + 1e-5f);
        const float mk = maskV[gr];
#pragma unroll
        for (int j = 0; j < 4; ++j)
            hvout[gr * NH + c0 + j] = mk * (gam[c0 + j] * (x[j] - mu) * rs + bet[c0 + j]);
    }
}

extern "C" void kernel_launch(void* const* d_in, const int* in_sizes, int n_in,
                              void* d_out, int out_size, void* d_ws, size_t ws_size,
                              hipStream_t stream) {
    (void)in_sizes; (void)n_in; (void)out_size; (void)ws_size;
    const float* h_V   = (const float*)d_in[0];
    const float* h_E   = (const float*)d_in[1];
    const int*   E_idx = (const int*)d_in[2];
    const float* maskV = (const float*)d_in[3];
    const float* maskA = (const float*)d_in[4];
    const float* W1  = (const float*)d_in[5];  const float* b1  = (const float*)d_in[6];
    const float* W2  = (const float*)d_in[7];  const float* b2  = (const float*)d_in[8];
    const float* W3  = (const float*)d_in[9];  const float* b3  = (const float*)d_in[10];
    const float* W11 = (const float*)d_in[11]; const float* b11 = (const float*)d_in[12];
    const float* W12 = (const float*)d_in[13]; const float* b12 = (const float*)d_in[14];
    const float* W13 = (const float*)d_in[15]; const float* b13 = (const float*)d_in[16];
    const float* Win = (const float*)d_in[17]; const float* bin_= (const float*)d_in[18];
    const float* Wout= (const float*)d_in[19]; const float* bout= (const float*)d_in[20];
    const float* g1  = (const float*)d_in[21]; const float* be1 = (const float*)d_in[22];
    const float* g2  = (const float*)d_in[23]; const float* be2 = (const float*)d_in[24];
    const float* g3  = (const float*)d_in[25]; const float* be3 = (const float*)d_in[26];

    const size_t HV = (size_t)NB * NN * NH;         // 262144
    const size_t HE = (size_t)NB * NN * NKN * NH;   // 12582912

    float* outV = (float*)d_out;                    // evolving h_V
    float* outE = outV + HV;                        // evolving h_E (in-place edge update)

    float*  hv_tmp = (float*)d_ws;                  // ping buffer for h_V after node update
    __bf16* wb = (__bf16*)((char*)d_ws + HV * sizeof(float));
    __bf16* W1t   = wb;                   // [3][128][384]
    __bf16* W2t   = W1t  + 3 * NH * H3;   // [3][128][128]
    __bf16* W3t   = W2t  + 3 * NH * NH;
    __bf16* W11t  = W3t  + 3 * NH * NH;
    __bf16* W12t  = W11t + 3 * NH * H3;
    __bf16* W13t  = W12t + 3 * NH * NH;
    __bf16* Wint  = W13t + 3 * NH * NH;   // [3][512][128]
    __bf16* Woutt = Wint + 3 * FF * NH;   // [3][128][512]

    convert_transpose<<<256, 256, 0, stream>>>(W1,  W1t,  H3, NH);
    convert_transpose<<<256, 256, 0, stream>>>(W2,  W2t,  NH, NH);
    convert_transpose<<<256, 256, 0, stream>>>(W3,  W3t,  NH, NH);
    convert_transpose<<<256, 256, 0, stream>>>(W11, W11t, H3, NH);
    convert_transpose<<<256, 256, 0, stream>>>(W12, W12t, NH, NH);
    convert_transpose<<<256, 256, 0, stream>>>(W13, W13t, NH, NH);
    convert_transpose<<<256, 256, 0, stream>>>(Win, Wint, NH, FF);
    convert_transpose<<<256, 256, 0, stream>>>(Wout,Woutt,FF, NH);

    hipMemcpyAsync(outV, h_V, HV * sizeof(float), hipMemcpyDeviceToDevice, stream);
    hipMemcpyAsync(outE, h_E, HE * sizeof(float), hipMemcpyDeviceToDevice, stream);

    for (int l = 0; l < 3; ++l) {
        // node message MLP + aggregate + LN1 -> hv_tmp
        msg_kernel<<<NB * NN, 256, 0, stream>>>(outV, outE, E_idx, maskA,
            W1t + (size_t)l * NH * H3, b1 + l * NH,
            W2t + (size_t)l * NH * NH, b2 + l * NH,
            W3t + (size_t)l * NH * NH, b3 + l * NH,
            g1 + l * NH, be1 + l * NH, hv_tmp, 0);
        // FFN + LN2 + mask_V -> outV
        ffn_kernel<<<NB * NN / 16, 256, 0, stream>>>(hv_tmp, maskV,
            Wint  + (size_t)l * FF * NH, bin_ + l * FF,
            Woutt + (size_t)l * NH * FF, bout + l * NH,
            g2 + l * NH, be2 + l * NH, outV);
        // edge message MLP + LN3 in place on outE
        msg_kernel<<<NB * NN, 256, 0, stream>>>(outV, outE, E_idx, maskA,
            W11t + (size_t)l * NH * H3, b11 + l * NH,
            W12t + (size_t)l * NH * NH, b12 + l * NH,
            W13t + (size_t)l * NH * NH, b13 + l * NH,
            g3 + l * NH, be3 + l * NH, outE, 1);
    }
}
